// PMP_45260365365872
// MI455X (gfx1250) — matrix operand order repaired and checked by reading, not compiled, not run on hardware
//
#include <hip/hip_runtime.h>

typedef __attribute__((ext_vector_type(16))) _Float16 v16h;
typedef __attribute__((ext_vector_type(8)))  float    v8f;

constexpr int  H_ = 256;
constexpr int  W_ = 1216;
constexpr int  B_ = 2;
constexpr long SS_ = (long)H_ * W_;     // 311296 = 2432*128  (tile-aligned)
constexpr long PP_ = (long)B_ * SS_;    // 622592 = 4864*128  (tile-aligned)

// ---------------------------------------------------------------------------
// Weight pre-pack: OIHW f32 -> fragment-order f16, layout
//   Wf[kk][nt][lane][e]  (512 f16 per (kk,nt) tile, 32B per lane)
// B-fragment mapping (16-bit B 32x16): lane l -> n = nt*16 + (l&15),
//   kcol = ((l>>4)<<4) + e;  k = kk*32+kcol; pos = k/Cinp; ci = k%Cinp.
// ---------------------------------------------------------------------------
__global__ __launch_bounds__(256)
void pack_wf(const float* __restrict__ Wt, _Float16* __restrict__ Wf,
             int Cin, int Cinp, int Cout, int NNT, int nK)
{
    long idx = (long)blockIdx.x * 256 + threadIdx.x;
    long total = (long)nK * NNT * 512;
    if (idx >= total) return;
    int kk   = (int)(idx / ((long)NNT * 512));
    int rem  = (int)(idx - (long)kk * NNT * 512);
    int nt   = rem >> 9;
    int le   = rem & 511;
    int lane = le >> 4;
    int e    = le & 15;
    int n    = (nt << 4) + (lane & 15);
    int kcol = ((lane >> 4) << 4) + e;
    int k    = (kk << 5) + kcol;
    int pos  = k / Cinp;
    int ci   = k - pos * Cinp;
    float v  = 0.f;
    if (n < Cout && ci < Cin) v = Wt[((long)n * Cin + ci) * 9 + pos];
    Wf[idx] = (_Float16)v;
}

// Compile-time WMMA chain: reuse_a hint on all but the last of the
// back-to-back identical WMMAs that share the A fragment (ISA 7.12 RA).
template <int NT, int NNT>
struct WmmaChain {
    static __device__ __forceinline__ void run(v16h af, const v16h* bf, v8f* acc) {
        acc[NT] = __builtin_amdgcn_wmma_f32_16x16x32_f16(
            false, af, false, bf[NT], (short)0, acc[NT],
            /*reuse_a=*/(NT + 1 < NNT), /*reuse_b=*/false);
        WmmaChain<NT + 1, NNT>::run(af, bf, acc);
    }
};
template <int NNT>
struct WmmaChain<NNT, NNT> {
    static __device__ __forceinline__ void run(v16h, const v16h*, v8f*) {}
};

// ---------------------------------------------------------------------------
// Implicit-GEMM 3x3 conv (stride 1, SAME) using WMMA f32_16x16x32_f16.
// GEMM: M = 128-pixel tile (8 waves x 16), N = Cout (NNT x 16), K = 9*CINP.
// Tap-major K: k = (ky*3+kx)*CINP + ci, CINP multiple of 32 -> each 32-wide
// K chunk is one conv tap: one border check per chunk, 16 SS_-strided loads.
// A: LDS double-buffered in fragment order (one 32B ds read per lane).
// B: loaded directly from the packed f16 global buffer (L2-resident).
// ---------------------------------------------------------------------------
template <int CIN, int CINP, int NNT>
__global__ __launch_bounds__(256)
void conv3x3_wmma(const float* __restrict__ X, const _Float16* __restrict__ Wf,
                  const float* __restrict__ scale, const float* __restrict__ bias,
                  float* __restrict__ Out, int Cout, int do_relu)
{
    constexpr int Keff = 9 * CINP;
    constexpr int nK   = Keff / 32;        // exact
    constexpr int CPT  = CINP / 32;        // chunks per tap (1 or 2)

    __shared__ __align__(32) _Float16 lAf[2][8][32][16];

    const int t    = threadIdx.x;
    const int lane = t & 31;
    const int wave = t >> 5;
    const long tile0 = (long)blockIdx.x * 128;   // always a full tile
    const int  bb    = (int)(tile0 / SS_);       // uniform per block
    const int  rbase = (int)(tile0 - (long)bb * SS_);

    // A staging ownership: pixel mloc = t>>1, k-half = t&1 (16 ci's each)
    const int mloc  = t >> 1;
    const int r0    = rbase + mloc;
    const int y     = r0 / W_;
    const int x     = r0 - y * W_;
    const int kbase = (t & 1) << 4;
    const float* Xb = X + (long)bb * CIN * SS_;

    float aReg[16];

    auto fetchA = [&](int kk) {
        int pos   = kk / CPT;                          // uniform tap index
        int cbase = (kk - pos * CPT) * 32 + kbase;
        int ky = pos / 3, kx = pos - (pos / 3) * 3;    // uniform
        int yy = y + ky - 1, xx = x + kx - 1;
        bool ok = ((unsigned)yy < (unsigned)H_) && ((unsigned)xx < (unsigned)W_);
        const float* src = Xb + (long)cbase * SS_ + (long)yy * W_ + xx;
        if (ok) {
            #pragma unroll
            for (int j = 0; j < 16; ++j) {
                float v = 0.f;
                if (CIN == CINP || (cbase + j) < CIN) v = src[(long)j * SS_];
                aReg[j] = v;
            }
        } else {
            #pragma unroll
            for (int j = 0; j < 16; ++j) aReg[j] = 0.f;
        }
    };
    auto storeA = [&](int pb) {
        #pragma unroll
        for (int j = 0; j < 16; ++j) {
            int kcol = kbase + j;
            int ld   = (mloc & 15) + ((kcol >> 4) << 4);
            int e    = kcol & 15;
            lAf[pb][mloc >> 4][ld][e] = (_Float16)aReg[j];
        }
    };

    v8f zero = {0.f, 0.f, 0.f, 0.f, 0.f, 0.f, 0.f, 0.f};
    v8f acc[NNT];
    #pragma unroll
    for (int nt = 0; nt < NNT; ++nt) acc[nt] = zero;

    const _Float16* wfl = Wf + (lane << 4);   // per-lane 32B slice

    fetchA(0);
    storeA(0);
    for (int kk = 0; kk < nK; ++kk) {
        __syncthreads();
        const int cb = kk & 1;
        v16h af = *reinterpret_cast<const v16h*>(&lAf[cb][wave][lane][0]);
        v16h bf[NNT];
        #pragma unroll
        for (int nt = 0; nt < NNT; ++nt)
            bf[nt] = *reinterpret_cast<const v16h*>(
                wfl + (((long)kk * NNT + nt) << 9));
        if (kk + 1 < nK) fetchA(kk + 1);
        WmmaChain<0, NNT>::run(af, bf, acc);
        if (kk + 1 < nK) storeA(cb ^ 1);
    }

    // epilogue: 8 consecutive pixels per lane -> vectorizable stores
    #pragma unroll
    for (int nt = 0; nt < NNT; ++nt) {
        int co = (nt << 4) + (lane & 15);
        if (co < Cout) {
            float sc = scale ? scale[co] : 1.f;
            float bi = bias  ? bias[co]  : 0.f;
            long obase = ((long)(bb * Cout + co)) * SS_ + rbase +
                         (wave << 4) + ((lane >> 4) << 3);
            #pragma unroll
            for (int r = 0; r < 8; ++r) {
                float v = acc[nt][r] * sc + bi;
                if (do_relu) v = fmaxf(v, 0.f);
                Out[obase + r] = v;
            }
        }
    }
}

// ---------------------------------------------------------------------------
// GenKernel normalization: w = w / (sum|w| + eps); mid = 1 - sum(w);
// output channels: [0..half-1]=w[0..half-1], [half]=mid, [half+1..co]=w[half..]
// ---------------------------------------------------------------------------
__global__ __launch_bounds__(256)
void gen_norm(const float* __restrict__ wraw, float* __restrict__ wout, int co)
{
    long p = (long)blockIdx.x * 256 + threadIdx.x;
    if (p >= PP_) return;
    int  b = (int)(p / SS_);
    long r = p - (long)b * SS_;
    const float* src = wraw + (long)b * co * SS_ + r;
    float*       dst = wout + (long)b * (co + 1) * SS_ + r;
    float sa = 0.f;
    for (int c = 0; c < co; ++c) sa += fabsf(src[(long)c * SS_]);
    float inv = 1.f / (sa + 1e-6f);
    int half = co >> 1;
    float s = 0.f;
    for (int c = 0; c < co; ++c) {
        float wn = src[(long)c * SS_] * inv;
        s += wn;
        dst[(long)((c < half) ? c : c + 1) * SS_] = wn;
    }
    dst[(long)half * SS_] = 1.f - s;
}

// sigmoid(raw) * (h0 > 0.001)   over [B,3,H,W]
__global__ __launch_bounds__(256)
void mask_sigmoid(const float* __restrict__ raw, const float* __restrict__ h0,
                  float* __restrict__ mask)
{
    long i = (long)blockIdx.x * 256 + threadIdx.x;
    long total = 3L * PP_;
    if (i >= total) return;
    long per = 3L * SS_;
    int  b  = (int)(i / per);
    long rr = i - (long)b * per;
    long r  = rr % SS_;
    float v  = raw[i];
    float sg = 1.f / (1.f + __expf(-v));
    float va = (h0[(long)b * SS_ + r] > 0.001f) ? 1.f : 0.f;
    mask[i] = sg * va;
}

// channel softmax over 3 channels, per pixel
__global__ __launch_bounds__(256)
void softmax3(const float* __restrict__ raw, float* __restrict__ out)
{
    long p = (long)blockIdx.x * 256 + threadIdx.x;
    if (p >= PP_) return;
    int  b = (int)(p / SS_);
    long r = p - (long)b * SS_;
    const float* s = raw + (long)b * 3 * SS_ + r;
    float a0 = s[0], a1 = s[SS_], a2 = s[2 * SS_];
    float mx = fmaxf(a0, fmaxf(a1, a2));
    float e0 = __expf(a0 - mx), e1 = __expf(a1 - mx), e2 = __expf(a2 - mx);
    float inv = 1.f / (e0 + e1 + e2);
    float* d = out + (long)b * 3 * SS_ + r;
    d[0]       = e0 * inv;
    d[SS_]     = e1 * inv;
    d[2 * SS_] = e2 * inv;
}

// h3 = h5 = h7 = hn ; hns channel 0 = hn
__global__ __launch_bounds__(256)
void init_h(const float* __restrict__ hn, float* __restrict__ hA,
            float* __restrict__ hns)
{
    long p = (long)blockIdx.x * 256 + threadIdx.x;
    if (p >= PP_) return;
    float v = hn[p];
    hA[p] = v; hA[p + PP_] = v; hA[p + 2 * PP_] = v;
    int  b = (int)(p / SS_);
    long r = p - (long)b * SS_;
    hns[(long)b * 3 * SS_ + r] = v;
}

// One CSPN propagation step for all three kernel sizes (zero padding).
__global__ __launch_bounds__(256)
void cspn_step(const float* __restrict__ hin, const float* __restrict__ w3,
               const float* __restrict__ w5, const float* __restrict__ w7,
               const float* __restrict__ mask, const float* __restrict__ h0,
               float* __restrict__ hout)
{
    long p = (long)blockIdx.x * 256 + threadIdx.x;
    if (p >= PP_) return;
    int  b = (int)(p / SS_);
    int  r = (int)(p - (long)b * SS_);
    int  y = r / W_, x = r - (r / W_) * W_;
    const float* h3 = hin;
    const float* h5 = hin + PP_;
    const float* h7 = hin + 2 * PP_;
    const long bi = (long)b * SS_;

    float a3 = 0.f;
    {
        const float* wp = w3 + (long)b * 9 * SS_ + r;
        int c = 0;
        for (int dy = -1; dy <= 1; ++dy)
            for (int dx = -1; dx <= 1; ++dx, ++c) {
                int yy = y + dy, xx = x + dx;
                float hv = (yy >= 0 && yy < H_ && xx >= 0 && xx < W_)
                               ? h3[bi + (long)yy * W_ + xx] : 0.f;
                a3 += wp[(long)c * SS_] * hv;
            }
    }
    float a5 = 0.f;
    {
        const float* wp = w5 + (long)b * 25 * SS_ + r;
        int c = 0;
        for (int dy = -2; dy <= 2; ++dy)
            for (int dx = -2; dx <= 2; ++dx, ++c) {
                int yy = y + dy, xx = x + dx;
                float hv = (yy >= 0 && yy < H_ && xx >= 0 && xx < W_)
                               ? h5[bi + (long)yy * W_ + xx] : 0.f;
                a5 += wp[(long)c * SS_] * hv;
            }
    }
    float a7 = 0.f;
    {
        const float* wp = w7 + (long)b * 49 * SS_ + r;
        int c = 0;
        for (int dy = -3; dy <= 3; ++dy)
            for (int dx = -3; dx <= 3; ++dx, ++c) {
                int yy = y + dy, xx = x + dx;
                float hv = (yy >= 0 && yy < H_ && xx >= 0 && xx < W_)
                               ? h7[bi + (long)yy * W_ + xx] : 0.f;
                a7 += wp[(long)c * SS_] * hv;
            }
    }
    float m3v = mask[(long)b * 3 * SS_ + r];
    float m5v = mask[(long)b * 3 * SS_ + SS_ + r];
    float m7v = mask[(long)b * 3 * SS_ + 2 * SS_ + r];
    float h0v = h0[p];
    hout[p]            = (1.f - m3v) * a3 + m3v * h0v;
    hout[p + PP_]      = (1.f - m5v) * a5 + m5v * h0v;
    hout[p + 2 * PP_]  = (1.f - m7v) * a7 + m7v * h0v;
}

// hns[:,ch] = c3*h3 + c5*h5 + c7*h7
__global__ __launch_bounds__(256)
void combo(const float* __restrict__ conf, const float* __restrict__ h,
           float* __restrict__ hns, int ch)
{
    long p = (long)blockIdx.x * 256 + threadIdx.x;
    if (p >= PP_) return;
    int  b = (int)(p / SS_);
    long r = p - (long)b * SS_;
    const float* c = conf + (long)b * 3 * SS_ + r;
    float v = c[0] * h[p] + c[SS_] * h[p + PP_] + c[2 * SS_] * h[p + 2 * PP_];
    hns[(long)b * 3 * SS_ + (long)ch * SS_ + r] = v;
}

// fcat = concat(fout[32ch], hns[3ch])  -> [B,35,H,W]
__global__ __launch_bounds__(256)
void pack_fcat(const float* __restrict__ fout, const float* __restrict__ hns,
               float* __restrict__ fcat)
{
    long i = (long)blockIdx.x * 256 + threadIdx.x;
    long per = 35L * SS_;
    long total = (long)B_ * per;
    if (i >= total) return;
    int  b  = (int)(i / per);
    long rr = i - (long)b * per;
    int  c  = (int)(rr / SS_);
    long r  = rr - (long)c * SS_;
    float v;
    if (c < 32) v = fout[((long)b * 32 + c) * SS_ + r];
    else        v = hns[((long)b * 3 + (c - 32)) * SS_ + r];
    fcat[i] = v;
}

// out = sum_ch wt[ch] * hns[ch]
__global__ __launch_bounds__(256)
void final_sum(const float* __restrict__ wt, const float* __restrict__ hns,
               float* __restrict__ out)
{
    long p = (long)blockIdx.x * 256 + threadIdx.x;
    if (p >= PP_) return;
    int  b = (int)(p / SS_);
    long r = p - (long)b * SS_;
    const float* w = wt  + (long)b * 3 * SS_ + r;
    const float* h = hns + (long)b * 3 * SS_ + r;
    out[p] = w[0] * h[0] + w[SS_] * h[SS_] + w[2 * SS_] * h[2 * SS_];
}

// ---------------------------------------------------------------------------
static void launch_conv(const float* X, const float* Wt, const float* sc,
                        const float* bi, float* Out, _Float16* wpack,
                        int Cin, int Cout, int relu, hipStream_t s)
{
    const int Cinp = (Cin == 32) ? 32 : 64;
    const int NNT  = (Cout + 15) / 16;
    const int nK   = 9 * Cinp / 32;
    const long tot = (long)nK * NNT * 512;
    pack_wf<<<(unsigned)((tot + 255) / 256), 256, 0, s>>>(
        Wt, wpack, Cin, Cinp, Cout, NNT, nK);

    const unsigned g = (unsigned)(PP_ / 128);
    if (Cin == 32) {
        if (Cout <= 16)
            conv3x3_wmma<32, 32, 1><<<g, 256, 0, s>>>(X, wpack, sc, bi, Out, Cout, relu);
        else if (Cout <= 32)
            conv3x3_wmma<32, 32, 2><<<g, 256, 0, s>>>(X, wpack, sc, bi, Out, Cout, relu);
        else
            conv3x3_wmma<32, 32, 3><<<g, 256, 0, s>>>(X, wpack, sc, bi, Out, Cout, relu);
    } else { // Cin == 35 (t-branch), Cout == 32; pad ci to 64 per tap
        conv3x3_wmma<35, 64, 2><<<g, 256, 0, s>>>(X, wpack, sc, bi, Out, Cout, relu);
    }
}

extern "C" void kernel_launch(void* const* d_in, const int* in_sizes, int n_in,
                              void* d_out, int out_size, void* d_ws, size_t ws_size,
                              hipStream_t stream) {
    const float* fout = (const float*)d_in[0];
    const float* hn   = (const float*)d_in[1];
    const float* h0   = (const float*)d_in[2];
    const float* k3w1 = (const float*)d_in[3];
    const float* k3s1 = (const float*)d_in[4];
    const float* k3b1 = (const float*)d_in[5];
    const float* k3w2 = (const float*)d_in[6];
    const float* k3s2 = (const float*)d_in[7];
    const float* k3b2 = (const float*)d_in[8];
    const float* k5w1 = (const float*)d_in[9];
    const float* k5s1 = (const float*)d_in[10];
    const float* k5b1 = (const float*)d_in[11];
    const float* k5w2 = (const float*)d_in[12];
    const float* k5s2 = (const float*)d_in[13];
    const float* k5b2 = (const float*)d_in[14];
    const float* k7w1 = (const float*)d_in[15];
    const float* k7s1 = (const float*)d_in[16];
    const float* k7b1 = (const float*)d_in[17];
    const float* k7w2 = (const float*)d_in[18];
    const float* k7s2 = (const float*)d_in[19];
    const float* k7b2 = (const float*)d_in[20];
    const float* mw1  = (const float*)d_in[21];
    const float* ms1  = (const float*)d_in[22];
    const float* mb1  = (const float*)d_in[23];
    const float* mw2  = (const float*)d_in[24];
    const float* mbias2 = (const float*)d_in[25];
    const float* cw1  = (const float*)d_in[26];
    const float* cs1  = (const float*)d_in[27];
    const float* cb1  = (const float*)d_in[28];
    const float* cw2  = (const float*)d_in[29];
    const float* cbias2 = (const float*)d_in[30];
    const float* tw1  = (const float*)d_in[31];
    const float* ts1  = (const float*)d_in[32];
    const float* tb1  = (const float*)d_in[33];
    const float* tw2  = (const float*)d_in[34];
    const float* tbias2 = (const float*)d_in[35];
    float* out = (float*)d_out;

    float* ws = (float*)d_ws;
    const long P = PP_;
    float* htmp  = ws;               // 32P
    float* wraw  = htmp  + 32 * P;   // 48P
    float* w3    = wraw  + 48 * P;   // 9P
    float* w5    = w3    + 9 * P;    // 25P
    float* w7    = w5    + 25 * P;   // 49P
    float* maskb = w7    + 49 * P;   // 3P
    float* confb = maskb + 3 * P;    // 3P
    float* hA    = confb + 3 * P;    // 3P
    float* hB    = hA    + 3 * P;    // 3P
    float* hns   = hB    + 3 * P;    // 3P
    float* fcat  = hns   + 3 * P;    // 35P
    float* wtb   = fcat  + 35 * P;   // 3P
    _Float16* wpack = (_Float16*)(wtb + 3 * P);   // <= 64K f16 (128KB region)

    const unsigned gP   = (unsigned)((PP_ + 255) / 256);
    const unsigned g3P  = (unsigned)((3 * PP_ + 255) / 256);
    const unsigned g35P = (unsigned)((35 * PP_ + 255) / 256);

    // k3 branch
    launch_conv(fout, k3w1, k3s1, k3b1, htmp, wpack, 32, 32, 1, stream);
    launch_conv(htmp, k3w2, k3s2, k3b2, wraw, wpack, 32, 8, 0, stream);
    gen_norm<<<gP, 256, 0, stream>>>(wraw, w3, 8);
    // k5 branch
    launch_conv(fout, k5w1, k5s1, k5b1, htmp, wpack, 32, 32, 1, stream);
    launch_conv(htmp, k5w2, k5s2, k5b2, wraw, wpack, 32, 24, 0, stream);
    gen_norm<<<gP, 256, 0, stream>>>(wraw, w5, 24);
    // k7 branch
    launch_conv(fout, k7w1, k7s1, k7b1, htmp, wpack, 32, 32, 1, stream);
    launch_conv(htmp, k7w2, k7s2, k7b2, wraw, wpack, 32, 48, 0, stream);
    gen_norm<<<gP, 256, 0, stream>>>(wraw, w7, 48);
    // mask branch
    launch_conv(fout, mw1, ms1, mb1, htmp, wpack, 32, 32, 1, stream);
    launch_conv(htmp, mw2, nullptr, mbias2, wraw, wpack, 32, 3, 0, stream);
    mask_sigmoid<<<g3P, 256, 0, stream>>>(wraw, h0, maskb);
    // conf branch
    launch_conv(fout, cw1, cs1, cb1, htmp, wpack, 32, 32, 1, stream);
    launch_conv(htmp, cw2, nullptr, cbias2, wraw, wpack, 32, 3, 0, stream);
    softmax3<<<gP, 256, 0, stream>>>(wraw, confb);

    // CSPN iterations
    init_h<<<gP, 256, 0, stream>>>(hn, hA, hns);
    float* cur = hA;
    float* nxt = hB;
    for (int i = 0; i < 6; ++i) {
        cspn_step<<<gP, 256, 0, stream>>>(cur, w3, w5, w7, maskb, h0, nxt);
        if (i == 2) combo<<<gP, 256, 0, stream>>>(confb, nxt, hns, 1);
        if (i == 5) combo<<<gP, 256, 0, stream>>>(confb, nxt, hns, 2);
        float* tmp = cur; cur = nxt; nxt = tmp;
    }

    // final attention over hns
    pack_fcat<<<g35P, 256, 0, stream>>>(fout, hns, fcat);
    launch_conv(fcat, tw1, ts1, tb1, htmp, wpack, 35, 32, 1, stream);
    launch_conv(htmp, tw2, nullptr, tbias2, wraw, wpack, 32, 3, 0, stream);
    softmax3<<<gP, 256, 0, stream>>>(wraw, wtb);
    final_sum<<<gP, 256, 0, stream>>>(wtb, hns, out);
}
